// SeqSeqModel_23828478558761
// MI455X (gfx1250) — compile-verified, matrix operand order checked by reading, not run on hardware
//
#include <hip/hip_runtime.h>
#include <hip/hip_bf16.h>

// Problem constants from the reference.
#define Bv 4
#define Tv 64
#define Nv 4096
#define Hv 128
#define Ev 65536
#define ROWS (Bv * Nv)             // 16384 rows per timestep GEMM
#define BNH ((long)Bv * Nv * Hv)   // 2,097,152
#define BN3H ((long)Bv * Nv * 3 * Hv)

typedef __attribute__((ext_vector_type(16))) __bf16 v16bf;
typedef __attribute__((ext_vector_type(8)))  float  v8f;

union ABfrag { v16bf v; unsigned int u[8]; };

// f32 pair -> packed bf16 dword: +0x8000 round bias, then ONE v_perm_b32
// grabs the two high halves. (cvt_pk_bf16_f32 builtin is absent on this TC.)
__device__ __forceinline__ unsigned int pk_bf16(float a, float b) {
  unsigned int ua = __float_as_uint(a) + 0x8000u;
  unsigned int ub = __float_as_uint(b) + 0x8000u;
  return __builtin_amdgcn_perm(ub, ua, 0x07060302u);
}
__device__ __forceinline__ unsigned short bf16_bits(float a) {
  return (unsigned short)((__float_as_uint(a) + 0x8000u) >> 16);
}

// ---------------------------------------------------------------------------
// f32 -> packed bf16 (row-major copy), grid-stride over pairs.
// ---------------------------------------------------------------------------
__global__ __launch_bounds__(256) void f32_to_bf16_kernel(
    const float* __restrict__ src, unsigned int* __restrict__ dst, long npairs) {
  long i = (long)blockIdx.x * blockDim.x + threadIdx.x;
  long stride = (long)gridDim.x * blockDim.x;
  for (; i < npairs; i += stride) {
    float2 v = ((const float2*)src)[i];
    dst[i] = pk_bf16(v.x, v.y);
  }
}

// ---------------------------------------------------------------------------
// Pack fp32 weight [K x NC] into the wave32 WMMA bf16 B-operand layout.
// Lane l (l<16): col = n0+l,      K elems {kb+0..7, kb+16..23}
// Lane l (>=16): col = n0+(l-16), K elems {kb+8..15, kb+24..31}
// pair tid = (((ntile*nKb + kb) << 5) + lane)*8 + j
// ---------------------------------------------------------------------------
__global__ __launch_bounds__(256) void pack_weights_kernel(
    const float* __restrict__ W, unsigned int* __restrict__ Wp, int K, int NC) {
  long tid = (long)blockIdx.x * blockDim.x + threadIdx.x;
  long npairs = (long)K * NC / 2;
  if (tid >= npairs) return;
  int j    = (int)(tid & 7);
  int lane = (int)((tid >> 3) & 31);
  long rest = tid >> 8;
  int nKb  = K >> 5;
  int kb   = (int)(rest % nKb);
  int ntile = (int)(rest / nKb);
  int col  = ntile * 16 + (lane & 15);
  int half = lane >> 4;
  int e0 = 2 * j, e1 = 2 * j + 1;
  int k0 = kb * 32 + ((e0 < 8) ? (half * 8 + e0) : (16 + half * 8 + (e0 - 8)));
  int k1 = kb * 32 + ((e1 < 8) ? (half * 8 + e1) : (16 + half * 8 + (e1 - 8)));
  Wp[tid] = pk_bf16(W[(long)k0 * NC + col], W[(long)k1 * NC + col]);
}

// ---------------------------------------------------------------------------
// WMMA GEMM, compile-time K/NC: C[ROWS x NC] = A[ROWS x K] * W + bias.
// One wave owns a 16x64 strip (4 accumulators); K loop fully unrolls.
// Per K-step: issue A loads + ALL 8 B loads first, then the 4 WMMAs, so
// in-order load completion lets WMMA i wait on loadcnt<=2*(3-i) instead of
// draining the pipe before every WMMA.
// A_BF16: A is pre-packed bf16 (two b128 loads, NO conversion VALU);
// else f32 loads + v_perm packing. OUT_BF16: store C as bf16 (b16 stores).
// Concat-K (K=256): k<128 reads A0, k>=128 reads A1. Row r -> b=r>>12,
// n=r&4095, element offset = b*stride + n*128.
// ---------------------------------------------------------------------------
template <int K, int NC, bool A_BF16, bool OUT_BF16>
__global__ __launch_bounds__(128) void wmma_gemm_kernel(
    const void* __restrict__ A0, const void* __restrict__ A1,
    const unsigned int* __restrict__ Wp, const float* __restrict__ bias,
    void* __restrict__ Cv, long batch_stride0, long batch_stride1) {
  constexpr int nKb  = K / 32;
  constexpr int ngrp = NC / 64;                 // 16x64 strips per row-tile
  const int wave  = threadIdx.x >> 5;
  const int lane  = threadIdx.x & 31;
  const int gtile = blockIdx.x * 4 + wave;      // grid sized exactly; no guard
  const int mtile = gtile / ngrp;
  const int grp   = gtile % ngrp;
  const int m0    = mtile * 16;
  const int row   = m0 + (lane & 15);
  const int b     = row >> 12;                  // Nv = 4096
  const int n     = row & (Nv - 1);
  const int half8 = (lane >> 4) << 3;           // 0 or 8
  const int ntile0 = grp * 4;

  v8f acc[4] = {};
#pragma unroll
  for (int kb = 0; kb < nKb; ++kb) {
    const int kglob = kb * 32;
    const void* src = (kglob < 128) ? A0 : A1;
    const long bs   = (kglob < 128) ? batch_stride0 : batch_stride1;
    const long aoff = (long)b * bs + (long)n * 128 + (kglob & 127) + half8;

    // ---- issue A loads ----
    ABfrag af;
    if constexpr (A_BF16) {
      const unsigned short* ap = (const unsigned short*)src + aoff;
      uint4 l0 = *(const uint4*)ap;             // K = base+0..7
      uint4 l1 = *(const uint4*)(ap + 16);      // K = base+16..23
      af.u[0] = l0.x; af.u[1] = l0.y; af.u[2] = l0.z; af.u[3] = l0.w;
      af.u[4] = l1.x; af.u[5] = l1.y; af.u[6] = l1.z; af.u[7] = l1.w;
    } else {
      const float4* ap4 = (const float4*)((const float*)src + aoff);
      float4 a0 = ap4[0], a1 = ap4[1], a2 = ap4[4], a3 = ap4[5];
      af.u[0] = pk_bf16(a0.x, a0.y); af.u[1] = pk_bf16(a0.z, a0.w);
      af.u[2] = pk_bf16(a1.x, a1.y); af.u[3] = pk_bf16(a1.z, a1.w);
      af.u[4] = pk_bf16(a2.x, a2.y); af.u[5] = pk_bf16(a2.z, a2.w);
      af.u[6] = pk_bf16(a3.x, a3.y); af.u[7] = pk_bf16(a3.z, a3.w);
    }

    // ---- issue ALL B loads for this K-step before any WMMA ----
    ABfrag bf[4];
#pragma unroll
    for (int tt = 0; tt < 4; ++tt) {
      const unsigned int* wp =
          Wp + ((((long)((ntile0 + tt) * nKb + kb)) << 5) + lane) * 8;
      __builtin_prefetch(wp + 256, 0, 1);       // next K-step's weight block
      const uint4* wp4 = (const uint4*)wp;
      uint4 w0 = wp4[0], w1 = wp4[1];
      bf[tt].u[0] = w0.x; bf[tt].u[1] = w0.y;
      bf[tt].u[2] = w0.z; bf[tt].u[3] = w0.w;
      bf[tt].u[4] = w1.x; bf[tt].u[5] = w1.y;
      bf[tt].u[6] = w1.z; bf[tt].u[7] = w1.w;
    }

    // ---- WMMA chain (one A fragment, 4 N-tiles) ----
#pragma unroll
    for (int tt = 0; tt < 4; ++tt)
      acc[tt] = __builtin_amdgcn_wmma_f32_16x16x32_bf16(
          false, af.v, false, bf[tt].v, (short)0, acc[tt], false, false);
  }

  // Store: one base, compile-time immediate offsets (r*NC + tt*16).
  const int colbase = ntile0 * 16 + (lane & 15);
  if constexpr (OUT_BF16) {
    unsigned short* cb =
        (unsigned short*)Cv + (long)(m0 + half8) * NC + colbase;
#pragma unroll
    for (int tt = 0; tt < 4; ++tt) {
      const float bv = bias[colbase + tt * 16];
#pragma unroll
      for (int r = 0; r < 8; ++r)
        cb[r * NC + tt * 16] = bf16_bits(acc[tt][r] + bv);
    }
  } else {
    float* cb = (float*)Cv + (long)(m0 + half8) * NC + colbase;
#pragma unroll
    for (int tt = 0; tt < 4; ++tt) {
      const float bv = bias[colbase + tt * 16];
#pragma unroll
      for (int r = 0; r < 8; ++r)
        cb[r * NC + tt * 16] = acc[tt][r] + bv;
    }
  }
}

// ---------------------------------------------------------------------------
// agg[:, edge_dst] += m[:, edge_src]  (f32 atomics; 4 feats/thread, all 4 batches)
// ---------------------------------------------------------------------------
__global__ __launch_bounds__(256) void scatter_add_kernel(
    const float* __restrict__ msg, float* __restrict__ agg,
    const int* __restrict__ esrc, const int* __restrict__ edst, int E) {
  int tid = blockIdx.x * blockDim.x + threadIdx.x;
  int chunk = tid & 31;                 // 32 chunks of 4 floats = H
  int e = tid >> 5;
  if (e >= E) return;
  int s = esrc[e], d = edst[e];
  int h0 = chunk << 2;
#pragma unroll
  for (int b = 0; b < Bv; ++b) {
    const float* mp = msg + ((long)b * Nv + s) * Hv + h0;
    float* ap = agg + ((long)b * Nv + d) * Hv + h0;
    float4 v = *(const float4*)mp;
    atomicAdd(ap + 0, v.x); atomicAdd(ap + 1, v.y);
    atomicAdd(ap + 2, v.z); atomicAdd(ap + 3, v.w);
  }
}

// ---------------------------------------------------------------------------
// Fused GRU elementwise + bf16 state store + readout dot + masked-MSE accum.
// One wave32 per (b,n) row; 2 adjacent feature pairs per lane (b64 loads,
// packed b32 bf16 stores); wave32 shuffle reduction for the readout dot.
// ---------------------------------------------------------------------------
__global__ __launch_bounds__(256) void gru_out_kernel(
    const float* __restrict__ gi, const float* __restrict__ gh,
    const float* __restrict__ xt, long x_batch_stride,
    unsigned int* __restrict__ statebf,
    const float* __restrict__ W_ro, const float* __restrict__ b_ro,
    const float* __restrict__ targets, const unsigned char* __restrict__ tmask,
    int t, float* __restrict__ acc) {
  int wid = blockIdx.x * (blockDim.x >> 5) + (threadIdx.x >> 5);
  int lane = threadIdx.x & 31;
  if (wid >= ROWS) return;
  int b = wid >> 12, n = wid & (Nv - 1);
  long grow = (long)wid * (3 * Hv);
  long xoff = (long)b * x_batch_stride + (long)n * Hv;
  float part = 0.f;
#pragma unroll
  for (int j = 0; j < 2; ++j) {
    int h = 2 * lane + 64 * j;
    float2 ir = *(const float2*)(gi + grow + h);
    float2 iz = *(const float2*)(gi + grow + Hv + h);
    float2 inn = *(const float2*)(gi + grow + 2 * Hv + h);
    float2 hr = *(const float2*)(gh + grow + h);
    float2 hz = *(const float2*)(gh + grow + Hv + h);
    float2 hn = *(const float2*)(gh + grow + 2 * Hv + h);
    float2 xv = *(const float2*)(xt + xoff + h);
    float r0 = 1.f / (1.f + __expf(-(ir.x + hr.x)));
    float r1 = 1.f / (1.f + __expf(-(ir.y + hr.y)));
    float z0 = 1.f / (1.f + __expf(-(iz.x + hz.x)));
    float z1 = 1.f / (1.f + __expf(-(iz.y + hz.y)));
    float n0 = tanhf(inn.x + r0 * hn.x);
    float n1 = tanhf(inn.y + r1 * hn.y);
    float s0 = (1.f - z0) * n0 + z0 * xv.x;
    float s1 = (1.f - z1) * n1 + z1 * xv.y;
    statebf[(long)wid * (Hv / 2) + (h >> 1)] = pk_bf16(s0, s1);
    float2 wr = *(const float2*)(W_ro + h);
    part += s0 * wr.x + s1 * wr.y;
  }
#pragma unroll
  for (int sh = 16; sh >= 1; sh >>= 1) part += __shfl_xor(part, sh, 32);
  if (lane == 0) {
    long ti = ((long)b * Tv + t) * Nv + n;
    if (tmask[ti]) {
      float d = (part + b_ro[0]) - targets[ti];
      atomicAdd(&acc[0], 0.5f * d * d);
      atomicAdd(&acc[1], 1.0f);
    }
  }
}

__global__ __launch_bounds__(256) void zero_f32_kernel(float* __restrict__ p, long n) {
  long i = (long)blockIdx.x * blockDim.x + threadIdx.x;
  long stride = (long)gridDim.x * blockDim.x;
  for (; i < n; i += stride) p[i] = 0.f;
}

__global__ void finalize_kernel(const float* __restrict__ acc, float* __restrict__ out) {
  if (threadIdx.x == 0 && blockIdx.x == 0) out[0] = acc[0] / acc[1];
}

// ---------------------------------------------------------------------------
extern "C" void kernel_launch(void* const* d_in, const int* in_sizes, int n_in,
                              void* d_out, int out_size, void* d_ws, size_t ws_size,
                              hipStream_t stream) {
  const float* x       = (const float*)d_in[0];
  const float* targets = (const float*)d_in[1];
  const float* W_msg   = (const float*)d_in[2];
  const float* b_msg   = (const float*)d_in[3];
  const float* W_mix   = (const float*)d_in[4];
  const float* b_mix   = (const float*)d_in[5];
  const float* W_ih    = (const float*)d_in[6];
  const float* b_ih    = (const float*)d_in[7];
  const float* W_hh    = (const float*)d_in[8];
  const float* b_hh    = (const float*)d_in[9];
  const float* W_ro    = (const float*)d_in[10];
  const float* b_ro    = (const float*)d_in[11];
  const unsigned char* tmask = (const unsigned char*)d_in[12];
  const int* esrc      = (const int*)d_in[13];
  const int* edst      = (const int*)d_in[14];
  float* out = (float*)d_out;

  char* ws = (char*)d_ws;
  size_t off = 0;
  auto alloc = [&](size_t bytes) -> void* {
    void* p = ws + off;
    off += (bytes + 255) & ~(size_t)255;
    return p;
  };
  float* acc     = (float*)alloc(2 * sizeof(float));
  float* mbuf    = (float*)alloc(BNH * sizeof(float));          // m (f32)
  float* agg     = (float*)alloc(BNH * sizeof(float));          // agg (f32, atomics)
  unsigned int* aggbf   = (unsigned int*)alloc(BNH * 2);        // agg (bf16)
  unsigned int* statebf = (unsigned int*)alloc(BNH * 2);        // state (bf16)
  unsigned int* mibf    = (unsigned int*)alloc(BNH * 2);        // mi (bf16)
  float* gi      = (float*)alloc(BN3H * sizeof(float));
  float* gh      = (float*)alloc(BN3H * sizeof(float));
  unsigned int* Wp_msg = (unsigned int*)alloc((size_t)128 * 128 / 2 * 4);
  unsigned int* Wp_mix = (unsigned int*)alloc((size_t)256 * 128 / 2 * 4);
  unsigned int* Wp_ih  = (unsigned int*)alloc((size_t)128 * 384 / 2 * 4);
  unsigned int* Wp_hh  = (unsigned int*)alloc((size_t)128 * 384 / 2 * 4);

  // Pre-converted bf16 copy of x, if workspace allows (268 MB).
  const size_t xbf_bytes = (size_t)Bv * Tv * Nv * Hv * 2;
  unsigned short* xbf = nullptr;
  if (ws_size >= off + xbf_bytes) xbf = (unsigned short*)alloc(xbf_bytes);

  const long TNH = (long)Tv * Nv * Hv;   // x batch stride (elements)
  const long NH  = (long)Nv * Hv;        // ws buffer batch stride (elements)

  zero_f32_kernel<<<1, 64, 0, stream>>>(acc, 2);
  zero_f32_kernel<<<1024, 256, 0, stream>>>((float*)statebf, BNH / 2);
  pack_weights_kernel<<<(128 * 128 / 2 + 255) / 256, 256, 0, stream>>>(W_msg, Wp_msg, 128, 128);
  pack_weights_kernel<<<(256 * 128 / 2 + 255) / 256, 256, 0, stream>>>(W_mix, Wp_mix, 256, 128);
  pack_weights_kernel<<<(128 * 384 / 2 + 255) / 256, 256, 0, stream>>>(W_ih, Wp_ih, 128, 384);
  pack_weights_kernel<<<(128 * 384 / 2 + 255) / 256, 256, 0, stream>>>(W_hh, Wp_hh, 128, 384);
  if (xbf)
    f32_to_bf16_kernel<<<16384, 256, 0, stream>>>(
        x, (unsigned int*)xbf, (long)Bv * Tv * Nv * Hv / 2);

  // Exact grids: (ROWS/16) row-tiles * (NC/64) strips / 4 waves per block.
  constexpr int blocks_n128 = (ROWS / 16) * (128 / 64) / 4;   // 512
  constexpr int blocks_n384 = (ROWS / 16) * (384 / 64) / 4;   // 1536

  for (int t = 0; t < Tv; ++t) {
    const float* xt = x + (long)t * NH;
    // m = x_t @ W_msg + b_msg
    if (xbf) {
      const unsigned short* xbt = xbf + (long)t * NH;
      wmma_gemm_kernel<128, 128, true, false><<<blocks_n128, 128, 0, stream>>>(
          xbt, xbt, Wp_msg, b_msg, mbuf, TNH, TNH);
    } else {
      wmma_gemm_kernel<128, 128, false, false><<<blocks_n128, 128, 0, stream>>>(
          xt, xt, Wp_msg, b_msg, mbuf, TNH, TNH);
    }
    // agg = scatter_add(m); then bf16 copy for the mix GEMM A-operand
    zero_f32_kernel<<<2048, 256, 0, stream>>>(agg, BNH);
    scatter_add_kernel<<<(Ev * 32) / 256, 256, 0, stream>>>(mbuf, agg, esrc, edst, Ev);
    f32_to_bf16_kernel<<<4096, 256, 0, stream>>>(agg, aggbf, BNH / 2);
    // mi = [agg | state] @ W_mix + b_mix   (K=256 split A, bf16 in, bf16 out)
    wmma_gemm_kernel<256, 128, true, true><<<blocks_n128, 128, 0, stream>>>(
        aggbf, statebf, Wp_mix, b_mix, mibf, NH, NH);
    // gi = mi @ W_ih + b_ih ; gh = x_t @ W_hh + b_hh
    wmma_gemm_kernel<128, 384, true, false><<<blocks_n384, 128, 0, stream>>>(
        mibf, mibf, Wp_ih, b_ih, gi, NH, NH);
    if (xbf) {
      const unsigned short* xbt = xbf + (long)t * NH;
      wmma_gemm_kernel<128, 384, true, false><<<blocks_n384, 128, 0, stream>>>(
          xbt, xbt, Wp_hh, b_hh, gh, TNH, TNH);
    } else {
      wmma_gemm_kernel<128, 384, false, false><<<blocks_n384, 128, 0, stream>>>(
          xt, xt, Wp_hh, b_hh, gh, TNH, TNH);
    }
    // GRU + bf16 state update + readout + masked MSE accumulation
    gru_out_kernel<<<ROWS / 8, 256, 0, stream>>>(
        gi, gh, xt, TNH, statebf, W_ro, b_ro, targets, tmask, t, acc);
  }
  finalize_kernel<<<1, 32, 0, stream>>>(acc, out);
}
